// HexConvResBlock_35184372089406
// MI455X (gfx1250) — matrix operand order repaired and checked by reading, not compiled, not run on hardware
//
#include <hip/hip_runtime.h>

// ---------------- types ----------------
typedef __bf16 bf16_t;
typedef __attribute__((ext_vector_type(16))) __bf16 v16bf;
typedef __attribute__((ext_vector_type(8)))  __bf16 v8bf;
typedef __attribute__((ext_vector_type(8)))  float  v8f;

#define POS        165           // positions per image (11*15)
#define CH         128           // channels
#define K7         896           // 7*CH
#define GRID_ROWS  221           // 13*17 zero-padded hex grid
#define LDS_STRIDE 136           // halves per row (128 + 8 pad -> bank spread)
#define ROW_BYTES  272           // LDS_STRIDE * sizeof(bf16)
#define MT         11            // M tiles (176 = 11*16 >= 165)
#define KB         28            // K blocks (896/32)
#define WELEMS     (K7*CH)       // 114688 bf16 per weight matrix
#define LDS_BYTES  (GRID_ROWS*ROW_BYTES)   // 60112, divisible by 16

// hex neighbor row offsets in the 13x17 grid, PRE-SCALED to LDS row bytes (x272)
__constant__ int EOFFB[7] = {-17*ROW_BYTES,-16*ROW_BYTES,-ROW_BYTES,0,ROW_BYTES,17*ROW_BYTES,18*ROW_BYTES}; // y1 even
__constant__ int OOFFB[7] = {-18*ROW_BYTES,-17*ROW_BYTES,-ROW_BYTES,0,ROW_BYTES,16*ROW_BYTES,17*ROW_BYTES}; // y1 odd

// ------------- prep: pack W (f32 [896,128]) into bf16 B-fragment layout -------------
// wp[c][ ((kb*8 + nt)*32 + lane)*16 + i ] = bf16( W_c[(kb*32 + (lane<16?0:16) + i)*128 + nt*16 + (lane&15)] )
__global__ void pack_w(const float* __restrict__ W1, const float* __restrict__ W2,
                       bf16_t* __restrict__ wp) {
    int tid = blockIdx.x * 256 + threadIdx.x;       // 0 .. 2*WELEMS-1
    int c   = tid / WELEMS;
    int r   = tid % WELEMS;
    int kb  = r >> 12;
    int r2  = r & 4095;
    int nt  = r2 >> 9;
    int r3  = r2 & 511;
    int lane= r3 >> 4;
    int i   = r3 & 15;
    int k   = kb * 32 + ((lane < 16) ? 0 : 16) + i;
    int n   = nt * 16 + (lane & 15);
    const float* W = c ? W2 : W1;
    wp[tid] = (bf16_t)W[k * CH + n];
}

// ------------- fused HexConvResBlock: one workgroup per image -------------
__global__ __launch_bounds__(256) void hexres(
    const float* __restrict__ x,
    const float* __restrict__ b1,
    const float* __restrict__ b2,
    const bf16_t* __restrict__ wp,
    float* __restrict__ out)
{
    __shared__ __align__(16) bf16_t lds[GRID_ROWS * LDS_STRIDE];   // 60112 B padded hex grid

    const int tid    = threadIdx.x;
    const int b      = blockIdx.x;
    const int lane   = tid & 31;
    const int wave   = tid >> 5;          // n-tile 0..7
    const int l15    = lane & 15;
    const int hiHalf = (lane >> 4) & 1;   // 0: lanes 0-15, 1: lanes 16-31
    const int klo    = hiHalf ? 16 : 0;   // A-fragment K sub-offset (halves)
    const int ncol   = wave * 16 + l15;   // this lane's output column

    // zero the padded grid with 16B stores (borders must be zero for the gather)
    {
        uint4 z; z.x = z.y = z.z = z.w = 0u;
        for (int i = tid; i < LDS_BYTES / 16; i += 256)
            ((uint4*)lds)[i] = z;
    }
    __syncthreads();

    // stage x image into LDS as bf16 at interior rows (2 channels per store)
    const float* xb = x + (size_t)b * (POS * CH);
    for (int i = tid; i < (POS * CH) / 2; i += 256) {
        float2 v = ((const float2*)xb)[i];
        int m   = i >> 6;            // / 64 pairs per row
        int cc2 = (i & 63) * 2;
        int y = m / 15, xx = m % 15;
        int row = (y + 1) * 17 + (xx + 1);
        union { bf16_t h[2]; unsigned int u; } pk;
        pk.h[0] = (bf16_t)v.x;
        pk.h[1] = (bf16_t)v.y;
        *(unsigned int*)((char*)lds + row * ROW_BYTES + cc2 * 2) = pk.u;
    }
    __syncthreads();

    // per-lane row BYTE bases for the 11 M-tiles (clamped; padded rows are discarded)
    int rowByte[MT]; int oddf[MT];
    #pragma unroll
    for (int mt = 0; mt < MT; ++mt) {
        int m = mt * 16 + l15; if (m > POS - 1) m = POS - 1;
        int y = m / 15, xx = m % 15;
        rowByte[mt] = ((y + 1) * 17 + (xx + 1)) * ROW_BYTES;
        oddf[mt]    = ((y & 1) == 0);        // y1 = y+1 odd -> odd-row offsets
    }

    v8f acc[MT];
    const char* lbase = (const char*)lds;

    auto conv = [&](const bf16_t* __restrict__ wpc, const float* __restrict__ bias) {
        float bv = bias[ncol];
        #pragma unroll
        for (int mt = 0; mt < MT; ++mt)
            #pragma unroll
            for (int r = 0; r < 8; ++r) acc[mt][r] = bv;

        #pragma unroll 2
        for (int kb = 0; kb < KB; ++kb) {
            int nb  = kb >> 2;                        // neighbor index 0..6 (uniform)
            int eob = EOFFB[nb], oob = OOFFB[nb];     // byte offsets (uniform scalars)
            int cbB = (kb & 3) * 64 + klo * 2;        // channel byte base within neighbor

            // B fragment: 32 contiguous bytes from packed weights
            union { v16bf v; v8bf h[2]; } bf;
            const v8bf* bp = (const v8bf*)(wpc + ((size_t)((kb * 8 + wave) * 32 + lane)) * 16);
            bf.h[0] = bp[0];
            bf.h[1] = bp[1];

            #pragma unroll
            for (int mt = 0; mt < MT; ++mt) {
                int abyte = rowByte[mt] + (oddf[mt] ? oob : eob) + cbB;   // cndmask + add
                union { v16bf v; v8bf h[2]; } af;
                af.h[0] = *(const v8bf*)(lbase + abyte);        // K chunk lo
                af.h[1] = *(const v8bf*)(lbase + abyte + 16);   // K chunk hi
                acc[mt] = __builtin_amdgcn_wmma_f32_16x16x32_bf16(
                    false, af.v, false, bf.v, (short)0, acc[mt], false, false);
            }
        }
    };

    // ---- conv1 ----
    conv(wp, b1);
    __syncthreads();   // everyone done reading x from LDS

    // LeakyReLU + write h back into the same padded grid (borders stay zero)
    #pragma unroll
    for (int mt = 0; mt < MT; ++mt) {
        #pragma unroll
        for (int r = 0; r < 8; ++r) {
            int m = mt * 16 + r + hiHalf * 8;       // C/D layout: M = r (+8 for hi lanes)
            if (m < POS) {
                float v = acc[mt][r];
                v = v > 0.f ? v : 0.01f * v;
                int y = m / 15, xx = m % 15;
                int row = (y + 1) * 17 + (xx + 1);
                lds[row * LDS_STRIDE + ncol] = (bf16_t)v;
            }
        }
    }
    __syncthreads();   // h visible to all waves

    // ---- conv2 ----
    conv(wp + WELEMS, b2);

    // ---- epilogue: residual add, f32 store ----
    float* ob = out + (size_t)b * (POS * CH);
    #pragma unroll
    for (int mt = 0; mt < MT; ++mt) {
        #pragma unroll
        for (int r = 0; r < 8; ++r) {
            int m = mt * 16 + r + hiHalf * 8;
            if (m < POS) {
                int idx = m * CH + ncol;
                ob[idx] = acc[mt][r] + xb[idx];
            }
        }
    }
}

// ------------- launch -------------
extern "C" void kernel_launch(void* const* d_in, const int* in_sizes, int n_in,
                              void* d_out, int out_size, void* d_ws, size_t ws_size,
                              hipStream_t stream) {
    const float* x  = (const float*)d_in[0];
    const float* W1 = (const float*)d_in[1];
    const float* b1 = (const float*)d_in[2];
    const float* W2 = (const float*)d_in[3];
    const float* b2 = (const float*)d_in[4];
    float* out = (float*)d_out;
    bf16_t* wp = (bf16_t*)d_ws;   // needs 2*114688*2 = 458752 bytes

    // pack W1,W2 -> bf16 fragment layout
    pack_w<<<(2 * WELEMS) / 256, 256, 0, stream>>>(W1, W2, wp);

    // one workgroup (8 waves) per batch image
    hexres<<<1024, 256, 0, stream>>>(x, b1, b2, wp, out);
}
